// LSTM_attention_71554155151406
// MI455X (gfx1250) — compile-verified, matrix operand order checked by reading, not run on hardware
//
#include <hip/hip_runtime.h>
#include <hip/hip_bf16.h>

#define HID 512
#define BATCH 64
#define TLEN 30
#define VOC 16000
#define FOURH 2048
#define NSPLIT 10                 // gridDim.y for logits GEMM
#define NTILES (VOC / 16)         // 1000
#define NT_PER_BLK (NTILES / NSPLIT) // 100

typedef __attribute__((ext_vector_type(16))) __bf16 v16bf;
typedef __attribute__((ext_vector_type(8)))  __bf16 v8bf;
typedef __attribute__((ext_vector_type(8)))  float  v8f;

// Load one 16x32 bf16 A/B fragment half-pair per the gfx1250 WMMA layout:
// elements 0..7  <- K = khalf + 0..7,  elements 8..15 <- K = khalf + 16..23
// (caller pre-offsets p by khalf = (lane>>4)*8).
__device__ __forceinline__ v16bf load_frag(const __bf16* p) {
    union { v16bf v; v8bf h[2]; } u;
    u.h[0] = *(const v8bf*)(p);
    u.h[1] = *(const v8bf*)(p + 16);
    return u.v;
}

// ---------------------------------------------------------------------------
// f32 -> bf16 conversion (for W, W_, V weights)
// ---------------------------------------------------------------------------
__global__ void k_f32_to_bf16(const float* __restrict__ src,
                              __bf16* __restrict__ dst, int n) {
    int i = blockIdx.x * 256 + threadIdx.x;
    if (i < n) dst[i] = (__bf16)src[i];
}

// ---------------------------------------------------------------------------
// Encoder + decoder LSTM recurrence, single workgroup of 32 waves.
// h is DOUBLE-BUFFERED in LDS (bf16, 2x64KB) so each task can run its matmul
// and gate epilogue back-to-back with no intra-step barrier and only one
// task's accumulators (4 x v8f = 32 VGPRs) live at a time -> no spills.
// A/B fragments are explicitly double-buffered across the K loop so L2/LDS
// load latency overlaps the 4 WMMAs of the previous K chunk.
// c stays in LDS as f32 (128KB). Total LDS = 256KB of the WGP's 320KB.
// ---------------------------------------------------------------------------
__global__ __launch_bounds__(1024, 1)
void k_lstm_recurrence(const int* __restrict__ x,
                       const int* __restrict__ y,
                       const float* __restrict__ U,
                       const float* __restrict__ Up,
                       const __bf16* __restrict__ Wbf,
                       const __bf16* __restrict__ Wpbf,
                       __bf16* __restrict__ Sbf) {
    extern __shared__ char smem[];
    __bf16* hbuf0 = (__bf16*)smem;                            // [64][512] bf16
    __bf16* hbuf1 = (__bf16*)(smem + BATCH * HID * 2);        // [64][512] bf16
    float*  c_lds = (float*)(smem + BATCH * HID * 4);         // [64][512] f32

    const int tid   = threadIdx.x;
    const int wave  = tid >> 5;
    const int lane  = tid & 31;
    const int lrow  = lane & 15;          // row/col within a 16x16 tile
    const int khalf = (lane >> 4) * 8;    // K half-offset (bf16 A/B layout)

    for (int i = tid; i < BATCH * HID; i += 1024) {
        hbuf0[i] = (__bf16)0.0f;
        c_lds[i] = 0.0f;
    }
    __syncthreads();

    for (int t = 0; t < 2 * TLEN; ++t) {
        const bool dec = (t >= TLEN);
        const int tt = dec ? (t - TLEN) : t;
        const __bf16* Wm = dec ? Wpbf : Wbf;   // (4H, H) bf16 row-major
        const float*  Um = dec ? Up   : U;     // (4H, VOC) f32 row-major
        const int*    ix = dec ? y    : x;     // (B, T) int
        const __bf16* hr = (t & 1) ? hbuf1 : hbuf0;   // read buffer
        __bf16*       hw = (t & 1) ? hbuf0 : hbuf1;   // write buffer

        for (int q = 0; q < 4; ++q) {
            const int task = wave + 32 * q;   // 0..127
            const int m = task >> 2;          // hidden-dim tile 0..31
            const int n = task & 3;           // batch tile 0..3
            const int bcol = n * 16 + lrow;   // batch column this lane owns

            const __bf16* hbase = hr + bcol * HID + khalf;
            int rowoff[4];
            #pragma unroll
            for (int g = 0; g < 4; ++g)
                rowoff[g] = ((g * 32 + m) * 16 + lrow) * HID + khalf;

            const v8f vzero = {};
            v8f acc[4] = {vzero, vzero, vzero, vzero};

            v16bf bfr[2];
            v16bf afr[2][4];
            bfr[0] = load_frag(hbase);
            #pragma unroll
            for (int g = 0; g < 4; ++g) afr[0][g] = load_frag(Wm + rowoff[g]);

            #pragma unroll
            for (int kc = 0; kc < 16; ++kc) {
                const int cur = kc & 1;
                const int nxt = cur ^ 1;
                if (kc < 15) {
                    const int kn = (kc + 1) * 32;
                    bfr[nxt] = load_frag(hbase + kn);
                    #pragma unroll
                    for (int g = 0; g < 4; ++g)
                        afr[nxt][g] = load_frag(Wm + rowoff[g] + kn);
                }
                #pragma unroll
                for (int g = 0; g < 4; ++g)
                    acc[g] = __builtin_amdgcn_wmma_f32_16x16x32_bf16(
                        false, afr[cur][g], false, bfr[cur], (short)0,
                        acc[g], false, false);
            }

            // Gate epilogue: fuse embedding gather U[:, token], apply gates,
            // update c (f32 LDS) and write h into the *other* buffer.
            const int token = ix[bcol * TLEN + tt];
            #pragma unroll
            for (int j = 0; j < 8; ++j) {
                // C/D layout: lanes 0-15 -> M=j, lanes 16-31 -> M=j+8
                const int r = m * 16 + khalf + j;   // 0..511 within one gate
                float vi = acc[0][j] + Um[(0 * HID + r) * VOC + token];
                float vf = acc[1][j] + Um[(1 * HID + r) * VOC + token];
                float vo = acc[2][j] + Um[(2 * HID + r) * VOC + token];
                float vg = acc[3][j] + Um[(3 * HID + r) * VOC + token];
                float gi = 1.0f / (1.0f + __expf(-vi));
                float gf = 1.0f / (1.0f + __expf(-vf));
                float go = 1.0f / (1.0f + __expf(-vo));
                float gg = tanhf(vg);
                float c_old = c_lds[bcol * HID + r];
                float c_new = gf * c_old + gi * gg;
                c_lds[bcol * HID + r] = c_new;
                float hv = go * tanhf(c_new);
                hw[bcol * HID + r] = (__bf16)hv;
                if (dec) Sbf[(tt * BATCH + bcol) * HID + r] = (__bf16)hv;
            }
        }
        __syncthreads();   // all hr reads + hw/c writes complete before swap
    }
}

// ---------------------------------------------------------------------------
// Batched logits GEMM: out(1920 x 16000) = S(1920 x 512) @ V^T, bf16 in /
// f32 WMMA accumulate. Grid = 120 M-tiles x 10 vocab chunks (1200 WGs).
// A fragments (16 K-chunks) live in registers, reused across 100 N tiles;
// B fragments (V rows, L2-resident) are double-buffered across K.
// ---------------------------------------------------------------------------
__global__ __launch_bounds__(256, 1)
void k_logits_gemm(const __bf16* __restrict__ Sbf,
                   const __bf16* __restrict__ Vbf,
                   float* __restrict__ out) {
    const int mtile = blockIdx.x;               // 0..119
    const int nbase = blockIdx.y * NT_PER_BLK;  // vocab tile base
    const int wave  = threadIdx.x >> 5;         // 0..7
    const int lane  = threadIdx.x & 31;
    const int lrow  = lane & 15;
    const int khalf = (lane >> 4) * 8;
    const int row   = mtile * 16 + lrow;        // 0..1919

    v16bf afrag[16];
    #pragma unroll
    for (int kc = 0; kc < 16; ++kc)
        afrag[kc] = load_frag(&Sbf[row * HID + kc * 32 + khalf]);

    for (int nt = nbase + wave; nt < nbase + NT_PER_BLK; nt += 8) {
        const __bf16* vb = &Vbf[(nt * 16 + lrow) * HID + khalf];
        if (nt + 8 < nbase + NT_PER_BLK)
            __builtin_prefetch(&Vbf[((nt + 8) * 16 + lrow) * HID], 0, 1);

        v8f acc = {};
        v16bf bfr[2];
        bfr[0] = load_frag(vb);
        #pragma unroll
        for (int kc = 0; kc < 16; ++kc) {
            const int cur = kc & 1;
            if (kc < 15) bfr[cur ^ 1] = load_frag(vb + (kc + 1) * 32);
            acc = __builtin_amdgcn_wmma_f32_16x16x32_bf16(
                false, afrag[kc], false, bfr[cur], (short)0, acc,
                false, false);
        }

        #pragma unroll
        for (int j = 0; j < 8; ++j) {
            const int orow = mtile * 16 + khalf + j;
            out[orow * VOC + nt * 16 + lrow] = acc[j];
        }
    }
}

// ---------------------------------------------------------------------------
// In-place row softmax over the 16000-wide vocab dimension.
// ---------------------------------------------------------------------------
__global__ void k_softmax_rows(float* __restrict__ out) {
    __shared__ float red[256];
    const int tid = threadIdx.x;
    float* p = out + (size_t)blockIdx.x * VOC;

    float m = -1e30f;
    for (int i = tid; i < VOC; i += 256) m = fmaxf(m, p[i]);
    red[tid] = m; __syncthreads();
    for (int s = 128; s > 0; s >>= 1) {
        if (tid < s) red[tid] = fmaxf(red[tid], red[tid + s]);
        __syncthreads();
    }
    m = red[0];
    __syncthreads();

    float sum = 0.0f;
    for (int i = tid; i < VOC; i += 256) sum += __expf(p[i] - m);
    red[tid] = sum; __syncthreads();
    for (int s = 128; s > 0; s >>= 1) {
        if (tid < s) red[tid] += red[tid + s];
        __syncthreads();
    }
    const float inv = 1.0f / red[0];

    for (int i = tid; i < VOC; i += 256) p[i] = __expf(p[i] - m) * inv;
}

// ---------------------------------------------------------------------------
// Launch: inputs are {x, y, U, W, U_, W_, V} per setup_inputs() order.
// ---------------------------------------------------------------------------
extern "C" void kernel_launch(void* const* d_in, const int* in_sizes, int n_in,
                              void* d_out, int out_size, void* d_ws, size_t ws_size,
                              hipStream_t stream) {
    const int*   x  = (const int*)d_in[0];
    const int*   y  = (const int*)d_in[1];
    const float* U  = (const float*)d_in[2];
    const float* W  = (const float*)d_in[3];
    const float* Up = (const float*)d_in[4];
    const float* Wp = (const float*)d_in[5];
    const float* V  = (const float*)d_in[6];
    float* out = (float*)d_out;

    // Workspace layout (bytes)
    char* ws = (char*)d_ws;
    const size_t nW  = (size_t)FOURH * HID;   // 1,048,576
    const size_t nV  = (size_t)VOC * HID;     // 8,192,000
    __bf16* Wbf  = (__bf16*)(ws);
    __bf16* Wpbf = (__bf16*)(ws + nW * 2);
    __bf16* Vbf  = (__bf16*)(ws + nW * 4);
    __bf16* Sbf  = (__bf16*)(ws + nW * 4 + nV * 2);

    // 1) Downconvert weights to bf16
    k_f32_to_bf16<<<(int)((nW + 255) / 256), 256, 0, stream>>>(W,  Wbf,  (int)nW);
    k_f32_to_bf16<<<(int)((nW + 255) / 256), 256, 0, stream>>>(Wp, Wpbf, (int)nW);
    k_f32_to_bf16<<<(int)((nV + 255) / 256), 256, 0, stream>>>(V,  Vbf,  (int)nV);

    // 2) Sequential encoder+decoder recurrence (single WGP, LDS-resident state)
    const size_t lds_bytes = (size_t)BATCH * HID * 2 * 2   // h double buffer
                           + (size_t)BATCH * HID * 4;      // c
    k_lstm_recurrence<<<1, 1024, lds_bytes, stream>>>(x, y, U, Up, Wbf, Wpbf, Sbf);

    // 3) Batched logits GEMM over all decoder steps
    dim3 ggrid((TLEN * BATCH) / 16, NSPLIT);
    k_logits_gemm<<<ggrid, 256, 0, stream>>>(Sbf, Vbf, out);

    // 4) Row softmax
    k_softmax_rows<<<TLEN * BATCH, 256, 0, stream>>>(out);

    (void)in_sizes; (void)n_in; (void)out_size; (void)ws_size;
}
